// SelfAttention_62697932587362
// MI455X (gfx1250) — matrix operand order repaired and checked by reading, not compiled, not run on hardware
//
#include <hip/hip_runtime.h>
#include <hip/hip_bf16.h>
#include <cstdint>
#include <cstddef>

// ---------------------------------------------------------------------------
// Problem constants (B,T,D,H) = (4, 2048, 1024, 16), HD = 64
// ---------------------------------------------------------------------------
constexpr int CB = 4;
constexpr int CT = 2048;
constexpr int CD = 1024;
constexpr int CH = 16;
constexpr int CHD = 64;

typedef __attribute__((ext_vector_type(16))) __bf16 v16bf;
typedef __attribute__((ext_vector_type(8)))  __bf16 v8bf;
typedef __attribute__((ext_vector_type(8)))  float  v8f;
typedef __attribute__((ext_vector_type(4)))  float  v4f;
typedef __attribute__((ext_vector_type(2)))  int    i32x2;
typedef __attribute__((ext_vector_type(4)))  int    i32x4;

__device__ inline v16bf cat8(v8bf lo, v8bf hi) {
  v16bf r;
#pragma unroll
  for (int i = 0; i < 8; ++i) { r[i] = lo[i]; r[i + 8] = hi[i]; }
  return r;
}

__device__ inline v8f wmma_bf16(v16bf a, v16bf b, v8f c) {
  return __builtin_amdgcn_wmma_f32_16x16x32_bf16(
      /*neg_a=*/false, a, /*neg_b=*/false, b,
      /*c_mod=*/(short)0, c, /*reuse_a=*/false, /*reuse_b=*/false);
}

// ---------------------------------------------------------------------------
// 16-lane butterfly reductions via DPP16 ROW_XMASK (pure VALU, no LDS traffic;
// replaces ds_bpermute that __shfl_xor lowers to). ROW_XMASK = 0x160 | mask,
// lane reads lane^mask within its row of 16 -> stays inside each half-wave,
// matching the C-matrix layout (row = vgpr_i + 8*(lane/16), col = lane%16).
// ---------------------------------------------------------------------------
template <int M>
__device__ inline float dpp_xor(float v) {
  int x = __builtin_amdgcn_update_dpp(0, __builtin_bit_cast(int, v),
                                      0x160 | M, 0xF, 0xF, true);
  return __builtin_bit_cast(float, x);
}
__device__ inline float row16_max(float v) {
  v = fmaxf(v, dpp_xor<1>(v));
  v = fmaxf(v, dpp_xor<2>(v));
  v = fmaxf(v, dpp_xor<4>(v));
  v = fmaxf(v, dpp_xor<8>(v));
  return v;
}
__device__ inline float row16_sum(float v) {
  v += dpp_xor<1>(v);
  v += dpp_xor<2>(v);
  v += dpp_xor<4>(v);
  v += dpp_xor<8>(v);
  return v;
}

// ---------------------------------------------------------------------------
// Async global->LDS copies (CDNA5 GLOBAL_LOAD_ASYNC_TO_LDS_*, ASYNCcnt).
// Probe-derived signature: (v{2,4}i addrspace(1)*, v{2,4}i addrspace(3)*,
// imm offset, imm cpol). Guarded with a synchronous fallback.
// ---------------------------------------------------------------------------
#if __has_builtin(__builtin_amdgcn_global_load_async_to_lds_b128) && \
    __has_builtin(__builtin_amdgcn_global_load_async_to_lds_b64)
#define HAVE_ASYNC_LDS 1
typedef __attribute__((address_space(1))) i32x2 ga_i32x2;
typedef __attribute__((address_space(3))) i32x2 ls_i32x2;
typedef __attribute__((address_space(1))) i32x4 ga_i32x4;
typedef __attribute__((address_space(3))) i32x4 ls_i32x4;
__device__ inline void async_b64(const __bf16* g, __bf16* l) {
  __builtin_amdgcn_global_load_async_to_lds_b64((ga_i32x2*)g, (ls_i32x2*)l, 0, 0);
}
__device__ inline void async_b128(const __bf16* g, __bf16* l) {
  __builtin_amdgcn_global_load_async_to_lds_b128((ga_i32x4*)g, (ls_i32x4*)l, 0, 0);
}
#else
#define HAVE_ASYNC_LDS 0
#endif

template <int N>
__device__ inline void wait_async() {
#if HAVE_ASYNC_LDS
#if __has_builtin(__builtin_amdgcn_s_wait_asynccnt)
  __builtin_amdgcn_s_wait_asynccnt((unsigned short)N);
#else
  if constexpr (N == 0) asm volatile("s_wait_asynccnt 0x0" ::: "memory");
  else                  asm volatile("s_wait_asynccnt 0x2" ::: "memory");
#endif
#endif
}

// ---------------------------------------------------------------------------
// Kernel 0/5: fp32 -> bf16 conversion (so GEMM staging can use async-LDS DMA,
// which cannot convert). ~50MB total traffic: ~2us at 23.3 TB/s, negligible.
// n must be a multiple of 8 (all our tensors are).
// ---------------------------------------------------------------------------
__global__ __launch_bounds__(256) void cvt_bf16_kernel(
    const float* __restrict__ src, __bf16* __restrict__ dst, int n) {
  const int i = (blockIdx.x * 256 + threadIdx.x) * 8;
  if (i >= n) return;
  v4f a = *(const v4f*)(src + i);
  v4f b = *(const v4f*)(src + i + 4);
  v8bf o;
#pragma unroll
  for (int k = 0; k < 4; ++k) { o[k] = (__bf16)a[k]; o[4 + k] = (__bf16)b[k]; }
  *(v8bf*)(dst + i) = o;
}

// ---------------------------------------------------------------------------
// Kernel 1/5 and 5/5: C[M,N] = A[M,K] * Bw[N,K]^T  (bf16 in, WMMA f32 accum)
//   WG tile 64(M) x 128(N), K-step 32, 8 waves; wave -> 16x64 (4 WMMA/step).
//   Double-buffered LDS filled by async global->LDS b64/b128 copies:
//   issue stage i+1, s_wait_asynccnt<=2 (own stage-i ops done, in-order),
//   barrier (all waves' data landed), compute, barrier, swap buffers.
// Fragment layouts used here (A/B contiguous k, same order in both; C per ISA):
//   A 16x32 bf16 : lane l -> row l%16 ; K 0..15 (l<16) or 16..31 (l>=16)
//   B 32x16 bf16 : lane l -> col l%16 ; K 0..15 (l<16) or 16..31 (l>=16)
//   C 16x16 f32  : vgpr i, lane l -> row i + 8*(l/16), col l%16
// ---------------------------------------------------------------------------
constexpr int LDT = 40;  // LDS row stride in bf16 (80B: 16B-aligned, conflict-free)

template <bool OUT_BF16>
__global__ __launch_bounds__(256) void gemm_wmma(
    const __bf16* __restrict__ A, const __bf16* __restrict__ Bw,
    void* __restrict__ Cout, int M, int N, int K) {
  __shared__ __bf16 As[2][64][LDT];
  __shared__ __bf16 Bs[2][128][LDT];

  const int tid  = threadIdx.x;
  const int lane = tid & 31;
  const int wave = tid >> 5;
  const int wm = wave & 3;      // 0..3 -> M rows
  const int wn = wave >> 2;     // 0..1 -> N cols
  const int lm = lane & 15;
  const int hi = lane >> 4;

  const int m0 = blockIdx.y * 64;
  const int n0 = blockIdx.x * 128;

  // per-thread staging assignment (all chunks 16B-aligned in LDS & global)
  const int ra = tid >> 2, ca = (tid & 3) * 8;    // A: 64x32, 8 elems (b64)
  const int rb = tid >> 1, cb = (tid & 1) * 16;   // B: 128x32, 16 elems (b128)
  const __bf16* gA = A + (size_t)(m0 + ra) * K + ca;
  const __bf16* gB = Bw + (size_t)(n0 + rb) * K + cb;

  auto stage = [&](int kk, int buf) {
#if HAVE_ASYNC_LDS
    async_b64(gA + kk, &As[buf][ra][ca]);
    async_b128(gB + kk, &Bs[buf][rb][cb]);
#else
    *(v8bf*)&As[buf][ra][ca] = *(const v8bf*)(gA + kk);
    *(v8bf*)&Bs[buf][rb][cb]      = *(const v8bf*)(gB + kk);
    *(v8bf*)&Bs[buf][rb][cb + 8]  = *(const v8bf*)(gB + kk + 8);
#endif
  };

  v8f acc[4] = {v8f{}, v8f{}, v8f{}, v8f{}};

  stage(0, 0);
  int buf = 0;
  for (int kk = 0; kk < K; kk += 32) {
    if (kk + 32 < K) { stage(kk + 32, buf ^ 1); wait_async<2>(); }
    else             { wait_async<0>(); }
    __syncthreads();

    const __bf16* ar = &As[buf][wm * 16 + lm][0];
    v16bf afr = cat8(*(const v8bf*)(ar + hi * 16),
                     *(const v8bf*)(ar + hi * 16 + 8));
#pragma unroll
    for (int j = 0; j < 4; ++j) {
      const __bf16* br = &Bs[buf][wn * 64 + j * 16 + lm][0];
      v16bf bfr = cat8(*(const v8bf*)(br + hi * 16),
                       *(const v8bf*)(br + hi * 16 + 8));
      acc[j] = wmma_bf16(afr, bfr, acc[j]);
    }
    __syncthreads();
    buf ^= 1;
  }

#pragma unroll
  for (int j = 0; j < 4; ++j) {
#pragma unroll
    for (int i = 0; i < 8; ++i) {
      const int row = m0 + wm * 16 + i + 8 * hi;
      const int col = n0 + wn * 64 + j * 16 + lm;
      if (OUT_BF16)
        ((__bf16*)Cout)[(size_t)row * N + col] = (__bf16)acc[j][i];
      else
        ((float*)Cout)[(size_t)row * N + col] = acc[j][i];
    }
  }
}

// ---------------------------------------------------------------------------
// Kernel 2/5: RoPE + layout transform.
//   qkv bf16 [B*T, 3*D] -> Qb,Kb bf16 [B,H,T,HD] (Q pre-scaled by 1/sqrt(HD))
//                          Vt bf16 [B,H,HD,T]    (transposed for PV B-frags)
// ---------------------------------------------------------------------------
__global__ __launch_bounds__(256) void rope_kernel(
    const __bf16* __restrict__ qkv, const float* __restrict__ rope,
    __bf16* __restrict__ Qb, __bf16* __restrict__ Kb, __bf16* __restrict__ Vt) {
  const int idx = blockIdx.x * 256 + threadIdx.x;
  const int i = idx & 31;
  const int h = (idx >> 5) & 15;
  const int t = (idx >> 9) & 2047;
  const int b = idx >> 20;

  const __bf16* qk = qkv + (size_t)(b * CT + t) * (3 * CD);
  const int col = h * CHD + 2 * i;
  const float c = rope[(t * 32 + i) * 2 + 0];
  const float s = rope[(t * 32 + i) * 2 + 1];

  const float q1 = (float)qk[col],          q2 = (float)qk[col + 1];
  const float k1 = (float)qk[CD + col],     k2 = (float)qk[CD + col + 1];
  const float v1 = (float)qk[2 * CD + col], v2 = (float)qk[2 * CD + col + 1];

  const size_t qkrow = ((size_t)(b * CH + h) * CT + t) * CHD;
  constexpr float kScale = 0.125f;  // 1/sqrt(64), folded into Q
  Qb[qkrow + 2 * i]     = (__bf16)((q1 * c - q2 * s) * kScale);
  Qb[qkrow + 2 * i + 1] = (__bf16)((q1 * s + q2 * c) * kScale);
  Kb[qkrow + 2 * i]     = (__bf16)(k1 * c - k2 * s);
  Kb[qkrow + 2 * i + 1] = (__bf16)(k1 * s + k2 * c);

  const size_t vtb = (size_t)(b * CH + h) * CHD;
  Vt[(vtb + 2 * i) * CT + t]     = (__bf16)v1;
  Vt[(vtb + 2 * i + 1) * CT + t] = (__bf16)v2;
}

// ---------------------------------------------------------------------------
// Kernel 3/5: causal flash attention (one wave per 16-query tile, 32-key
// blocks, online softmax with DPP row reductions, P re-shaped through a
// per-wave LDS slab, 8 WMMA per block). No block barriers in the loop (waves
// have different causal trip counts); EXEC stays all-ones around WMMA.
// ---------------------------------------------------------------------------
__global__ __launch_bounds__(256) void attn_kernel(
    const __bf16* __restrict__ Qb, const __bf16* __restrict__ Kb,
    const __bf16* __restrict__ Vt, __bf16* __restrict__ Ob) {
  __shared__ __bf16 pbuf[8][16][LDT];

  const int lane = threadIdx.x & 31;
  const int wave = threadIdx.x >> 5;
  const int lm = lane & 15;
  const int hi = lane >> 4;
  const int bh = blockIdx.y;
  const int q0 = (blockIdx.x * 8 + wave) * 16;

  const __bf16* Qbase = Qb + (size_t)bh * CT * CHD;
  const __bf16* Kbase = Kb + (size_t)bh * CT * CHD;
  const __bf16* Vbase = Vt + (size_t)bh * CHD * CT;

  v16bf qa[2];
  {
    const __bf16* qr = Qbase + (size_t)(q0 + lm) * CHD;
#pragma unroll
    for (int c = 0; c < 2; ++c)
      qa[c] = cat8(*(const v8bf*)(qr + c * 32 + hi * 16),
                   *(const v8bf*)(qr + c * 32 + hi * 16 + 8));
  }

  v8f oacc[4] = {v8f{}, v8f{}, v8f{}, v8f{}};
  float mrow[8], lrow[8];
#pragma unroll
  for (int i = 0; i < 8; ++i) { mrow[i] = -__builtin_inff(); lrow[i] = 0.0f; }

  const int qlast = q0 + 15;
  for (int k0 = 0; k0 <= qlast; k0 += 32) {
    // ---- S = Q * K^T for 32 keys: two 16x16 tiles ----
    v8f s[2];
#pragma unroll
    for (int st = 0; st < 2; ++st) {
      const __bf16* kr = Kbase + (size_t)(k0 + st * 16 + lm) * CHD;
      v8f a = {};
#pragma unroll
      for (int c = 0; c < 2; ++c) {
        v16bf kf = cat8(*(const v8bf*)(kr + c * 32 + hi * 16),
                        *(const v8bf*)(kr + c * 32 + hi * 16 + 8));
        a = wmma_bf16(qa[c], kf, a);
      }
      s[st] = a;
    }

    // ---- causal mask + online softmax (DPP butterfly reductions) ----
#pragma unroll
    for (int i = 0; i < 8; ++i) {
      const int q = q0 + i + 8 * hi;
      s[0][i] = (k0 + lm      > q) ? -__builtin_inff() : s[0][i];
      s[1][i] = (k0 + 16 + lm > q) ? -__builtin_inff() : s[1][i];
      const float mx = row16_max(fmaxf(s[0][i], s[1][i]));
      const float mnew  = fmaxf(mrow[i], mx);
      const float alpha = __expf(mrow[i] - mnew);
      const float p0 = __expf(s[0][i] - mnew);
      const float p1 = __expf(s[1][i] - mnew);
      const float rs = row16_sum(p0 + p1);
      lrow[i] = lrow[i] * alpha + rs;
      mrow[i] = mnew;
      s[0][i] = p0; s[1][i] = p1;
      oacc[0][i] *= alpha; oacc[1][i] *= alpha;
      oacc[2][i] *= alpha; oacc[3][i] *= alpha;
    }

    // ---- P: C-layout -> LDS -> A-layout (within this wave only) ----
    asm volatile("" ::: "memory");
#pragma unroll
    for (int i = 0; i < 8; ++i) {
      pbuf[wave][i + 8 * hi][lm]      = (__bf16)s[0][i];
      pbuf[wave][i + 8 * hi][16 + lm] = (__bf16)s[1][i];
    }
    asm volatile("s_wait_dscnt 0x0" ::: "memory");
    const __bf16* pr = &pbuf[wave][lm][0];
    v16bf pf = cat8(*(const v8bf*)(pr + hi * 16),
                    *(const v8bf*)(pr + hi * 16 + 8));
    asm volatile("" ::: "memory");

    // ---- O += P (16x32) * V (32x64): 4 WMMAs via transposed V ----
#pragma unroll
    for (int j = 0; j < 4; ++j) {
      const __bf16* vr = Vbase + (size_t)(j * 16 + lm) * CT + k0 + hi * 16;
      v16bf vf = cat8(*(const v8bf*)(vr), *(const v8bf*)(vr + 8));
      oacc[j] = wmma_bf16(pf, vf, oacc[j]);
    }
  }

  // ---- normalize + store O (bf16, [B*T, D]) ----
  const int b = bh >> 4, h = bh & 15;
#pragma unroll
  for (int i = 0; i < 8; ++i) {
    const float inv = 1.0f / lrow[i];
    const int t = q0 + i + 8 * hi;
    __bf16* orow = Ob + (size_t)(b * CT + t) * CD + h * CHD + lm;
    orow[0]  = (__bf16)(oacc[0][i] * inv);
    orow[16] = (__bf16)(oacc[1][i] * inv);
    orow[32] = (__bf16)(oacc[2][i] * inv);
    orow[48] = (__bf16)(oacc[3][i] * inv);
  }
}

// ---------------------------------------------------------------------------
// Launch: inputs = { x[f32], rope[f32], mask[bool,unused], W_attn[f32], W_proj[f32] }
// ws (bf16): qkv 48MB | Qb/Kb/Vt/Ob 16MB ea | Xb 16MB | Wab 6MB | Wpb 2MB ~ 143MB
// ---------------------------------------------------------------------------
extern "C" void kernel_launch(void* const* d_in, const int* in_sizes, int n_in,
                              void* d_out, int out_size, void* d_ws, size_t ws_size,
                              hipStream_t stream) {
  const float* x      = (const float*)d_in[0];
  const float* rope   = (const float*)d_in[1];
  const float* W_attn = (const float*)d_in[3];
  const float* W_proj = (const float*)d_in[4];
  float* out = (float*)d_out;

  const size_t MT = (size_t)CB * CT;          // 8192 rows
  __bf16* qkv = (__bf16*)d_ws;                // [MT, 3D]
  __bf16* Qb  = qkv + MT * (3 * CD);          // [B,H,T,HD]
  __bf16* Kb  = Qb + MT * CD;
  __bf16* Vt  = Kb + MT * CD;                 // [B,H,HD,T]
  __bf16* Ob  = Vt + MT * CD;                 // [MT, D]
  __bf16* Xb  = Ob + MT * CD;                 // [MT, D]
  __bf16* Wab = Xb + MT * CD;                 // [3D, D]
  __bf16* Wpb = Wab + (size_t)(3 * CD) * CD;  // [D, D]

  const int nx = (int)(MT * CD), na = 3 * CD * CD, np = CD * CD;
  cvt_bf16_kernel<<<nx / (256 * 8), 256, 0, stream>>>(x, Xb, nx);
  cvt_bf16_kernel<<<na / (256 * 8), 256, 0, stream>>>(W_attn, Wab, na);
  cvt_bf16_kernel<<<np / (256 * 8), 256, 0, stream>>>(W_proj, Wpb, np);

  // 1) qkv = x @ W_attn^T (M=8192, N=3072, K=1024) -> bf16
  gemm_wmma<true><<<dim3(3 * CD / 128, MT / 64), 256, 0, stream>>>(
      Xb, Wab, qkv, (int)MT, 3 * CD, CD);
  // 2) RoPE + transpose/split
  rope_kernel<<<(CB * CT * CH * 32) / 256, 256, 0, stream>>>(qkv, rope, Qb, Kb, Vt);
  // 3) causal flash attention
  attn_kernel<<<dim3(CT / 128, CB * CH), 256, 0, stream>>>(Qb, Kb, Vt, Ob);
  // 4) out = O @ W_proj^T (M=8192, N=1024, K=1024) -> fp32
  gemm_wmma<false><<<dim3(CD / 128, MT / 64), 256, 0, stream>>>(
      Ob, Wpb, out, (int)MT, CD, CD);
}
